// SetAbstraction_75084618268930
// MI455X (gfx1250) — matrix operand order repaired and checked by reading, not compiled, not run on hardware
//
#include <hip/hip_runtime.h>
#include <hip/hip_bf16.h>

#define B_    8
#define N1_   8192
#define F1_   64
#define N2_   1024
#define K_    32
#define IN_DIM_ 67
#define RAD2  0.04f
#define FLTMAX 3.402823466e+38f

// packed bf16 weight image: Wt0 (64x96) | Wt1 (64x64) | Wt2 (128x64), N-major
#define WPACK_ELEMS (64 * 96 + 64 * 64 + 128 * 64)   // 18432 (36,864 B)
#define WT1_OFF (64 * 96)
#define WT2_OFF (64 * 96 + 64 * 64)

typedef __attribute__((ext_vector_type(16))) __bf16        v16bf;
typedef __attribute__((ext_vector_type(8)))  float         v8f;
typedef __attribute__((ext_vector_type(4)))  unsigned int  v4u;
typedef __attribute__((ext_vector_type(4)))  unsigned int  u32x4;
typedef __attribute__((ext_vector_type(8)))  unsigned int  u32x8;

union FragU { v16bf bf; v4u u[2]; };

__device__ inline unsigned short f2bf(float f) {
  unsigned int u = __float_as_uint(f);
  return (unsigned short)((u + 0x7FFFu + ((u >> 16) & 1u)) >> 16);
}

__device__ inline v8f zero8() {
  v8f z = {0.f, 0.f, 0.f, 0.f, 0.f, 0.f, 0.f, 0.f};
  return z;
}

// A fragment (16x32 bf16, M = lane&15): K = i+8g (i<8), K = i+8+8g (i>=8), +32*ks
__device__ inline v16bf load_a_frag(const unsigned short* rowPtr, int ks, int g) {
  FragU f;
  const int c0 = ks * 32 + 8 * g;
  f.u[0] = *(const v4u*)(rowPtr + c0);
  f.u[1] = *(const v4u*)(rowPtr + c0 + 16);
  return f.bf;
}

// B fragment (32x16 bf16, N = lane&15): K = i+8g (i<8), K = i+8+8g (i>=8), +32*ks
__device__ inline v16bf load_b_frag(const unsigned short* rowPtr, int ks, int g) {
  FragU f;
  const int c0 = ks * 32 + 8 * g;
  f.u[0] = *(const v4u*)(rowPtr + c0);
  f.u[1] = *(const v4u*)(rowPtr + c0 + 16);
  return f.bf;
}

// TDM: one 1-D tile copy global -> LDS (18432 x 2B elements), 2-group descriptor
__device__ inline void tdm_load_weights(const unsigned short* gsrc, unsigned ldsDst) {
  unsigned long long ga = (unsigned long long)(uintptr_t)gsrc;
  const unsigned n = WPACK_ELEMS;
  u32x4 g0;
  g0[0] = 1u;                                              // count=1, user mode
  g0[1] = ldsDst;                                          // lds_addr
  g0[2] = (unsigned)(ga & 0xFFFFFFFFu);                    // global_addr[31:0]
  g0[3] = (unsigned)((ga >> 32) & 0x01FFFFFFu) | (2u << 30); // addr[56:32] | type=2
  u32x8 g1;
  g1[0] = (1u << 16);                  // workgroup_mask=0, data_size=1 (2 bytes)
  g1[1] = (n & 0xFFFFu) << 16;         // tensor_dim0[15:0]
  g1[2] = (n >> 16) | (1u << 16);      // tensor_dim0[31:16], tensor_dim1=1
  g1[3] = (n & 0xFFFFu) << 16;         // tensor_dim1 hi=0, tile_dim0 = n
  g1[4] = 0u;                          // tile_dim1=0, tile_dim2=0 (unused)
  g1[5] = n;                           // tensor_dim0_stride[31:0]
  g1[6] = 0u;                          // stride hi, tensor_dim1_stride lo
  g1[7] = 0u;
  asm volatile("tensor_load_to_lds %0, %1" :: "s"(g0), "s"(g1) : "memory");
}

// ---------------- Kernel 0: pack weights to bf16, N-major, padded -----------
// X row layout is [feat(0..63) | xyz(64..66) | pad(67..95)], weights permuted to match.
__global__ __launch_bounds__(256) void pack_weights_kernel(
    const float* __restrict__ W0, const float* __restrict__ W1,
    const float* __restrict__ W2, unsigned short* __restrict__ wp) {
  int i = blockIdx.x * 256 + threadIdx.x;
  if (i >= WPACK_ELEMS) return;
  unsigned short v;
  if (i < WT1_OFF) {
    int n = i / 96, c = i - n * 96;
    float f = 0.f;
    if (c < 64) f = W0[(3 + c) * 64 + n];        // feature channels
    else if (c < IN_DIM_) f = W0[(c - 64) * 64 + n]; // xyz channels
    v = f2bf(f);
  } else if (i < WT2_OFF) {
    int j = i - WT1_OFF; int n = j >> 6, c = j & 63;
    v = f2bf(W1[c * 64 + n]);
  } else {
    int j = i - WT2_OFF; int n = j >> 6, c = j & 63;
    v = f2bf(W2[c * 128 + n]);
  }
  wp[i] = v;
}

// ---------------- Kernel 1: farthest point sampling (1 block / batch) --------
__global__ __launch_bounds__(1024) void fps_kernel(const float* __restrict__ xyz,
                                                   float* __restrict__ selxyz) {
  const int b   = blockIdx.x;
  const int tid = threadIdx.x;
  const float* bx = xyz + (size_t)b * N1_ * 3;

  float px[8], py[8], pz[8], md[8];
#pragma unroll
  for (int j = 0; j < 8; ++j) {
    int p = tid + j * 1024;
    px[j] = bx[p * 3 + 0];
    py[j] = bx[p * 3 + 1];
    pz[j] = bx[p * 3 + 2];
    md[j] = FLTMAX;
  }

  __shared__ float sl[3];
  __shared__ float swv[32];
  __shared__ int   swi[32];

  if (tid == 0) {
    sl[0] = bx[0]; sl[1] = bx[1]; sl[2] = bx[2];
    float* o = selxyz + (size_t)b * N2_ * 3;
    o[0] = bx[0]; o[1] = bx[1]; o[2] = bx[2];
  }
  __syncthreads();

  const int wid = tid >> 5, lane = tid & 31;
  for (int t = 1; t < N2_; ++t) {
    float lx = sl[0], ly = sl[1], lz = sl[2];
    float bv = -1.0f; int bi = 0;
#pragma unroll
    for (int j = 0; j < 8; ++j) {
      float dx = px[j] - lx, dy = py[j] - ly, dz = pz[j] - lz;
      float d = dx * dx + dy * dy + dz * dz;
      md[j] = fminf(md[j], d);
      int p = tid + j * 1024;
      if (md[j] > bv || (md[j] == bv && p < bi)) { bv = md[j]; bi = p; }
    }
#pragma unroll
    for (int off = 16; off > 0; off >>= 1) {
      float ov = __shfl_down(bv, off, 32);
      int   oi = __shfl_down(bi, off, 32);
      if (ov > bv || (ov == bv && oi < bi)) { bv = ov; bi = oi; }
    }
    if (lane == 0) { swv[wid] = bv; swi[wid] = bi; }
    __syncthreads();
    if (tid == 0) {
      float gv = swv[0]; int gi = swi[0];
      for (int w = 1; w < 32; ++w)
        if (swv[w] > gv || (swv[w] == gv && swi[w] < gi)) { gv = swv[w]; gi = swi[w]; }
      float qx = bx[gi * 3 + 0], qy = bx[gi * 3 + 1], qz = bx[gi * 3 + 2];
      sl[0] = qx; sl[1] = qy; sl[2] = qz;
      float* o = selxyz + ((size_t)b * N2_ + t) * 3;
      o[0] = qx; o[1] = qy; o[2] = qz;
    }
    __syncthreads();
  }
}

// ---------------- Kernel 2: KNN top-32 with radius clamp (1 block / query) ---
__global__ __launch_bounds__(256) void knn_kernel(const float* __restrict__ xyz,
                                                  const float* __restrict__ selxyz,
                                                  int* __restrict__ nbr) {
  __shared__ float d2s[N1_];   // 32 KB
  __shared__ float swv[8];
  __shared__ int   swi[8];
  __shared__ int   sfirst;

  const int gq  = blockIdx.x;
  const int b   = gq >> 10;
  const int tid = threadIdx.x;
  const float* bx = xyz + (size_t)b * N1_ * 3;
  const float* Q  = selxyz + (size_t)gq * 3;
  const float qx = Q[0], qy = Q[1], qz = Q[2];

  for (int p = tid; p < N1_; p += 256) {
    float dx = bx[p * 3 + 0] - qx, dy = bx[p * 3 + 1] - qy, dz = bx[p * 3 + 2] - qz;
    d2s[p] = dx * dx + dy * dy + dz * dz;
  }
  __syncthreads();

  const int wid = tid >> 5, lane = tid & 31;
  for (int r = 0; r < K_; ++r) {
    float bv = FLTMAX; int bi = 0;
#pragma unroll 4
    for (int j = 0; j < N1_ / 256; ++j) {
      int p = tid + j * 256;
      float d = d2s[p];
      if (d < bv || (d == bv && p < bi)) { bv = d; bi = p; }
    }
#pragma unroll
    for (int off = 16; off > 0; off >>= 1) {
      float ov = __shfl_down(bv, off, 32);
      int   oi = __shfl_down(bi, off, 32);
      if (ov < bv || (ov == bv && oi < bi)) { bv = ov; bi = oi; }
    }
    if (lane == 0) { swv[wid] = bv; swi[wid] = bi; }
    __syncthreads();
    if (tid == 0) {
      float gv = swv[0]; int gi = swi[0];
      for (int w = 1; w < 8; ++w)
        if (swv[w] < gv || (swv[w] == gv && swi[w] < gi)) { gv = swv[w]; gi = swi[w]; }
      if (r == 0) sfirst = gi;
      nbr[(size_t)gq * K_ + r] = (r > 0 && gv > RAD2) ? sfirst : gi;
      d2s[gi] = FLTMAX;  // exclude from later rounds
    }
    __syncthreads();
  }
}

// ---------------- Kernel 3: gather + MLP (WMMA bf16) + max-pool --------------
template <int KSTEPS, int NTILES>
__device__ inline void layer_store(const unsigned short* Xw, int xStride,
                                   const unsigned short* Wt, int wStride,
                                   const float* __restrict__ bias,
                                   unsigned short* Ow, int oStride, int lane) {
  const int g = lane >> 4, l15 = lane & 15;
  v8f acc[2][NTILES];
#pragma unroll
  for (int mt = 0; mt < 2; ++mt)
#pragma unroll
    for (int nt = 0; nt < NTILES; ++nt) acc[mt][nt] = zero8();

#pragma unroll
  for (int ks = 0; ks < KSTEPS; ++ks) {
    v16bf a0 = load_a_frag(Xw + (0 * 16 + l15) * xStride, ks, g);
    v16bf a1 = load_a_frag(Xw + (1 * 16 + l15) * xStride, ks, g);
#pragma unroll
    for (int nt = 0; nt < NTILES; ++nt) {
      v16bf bf = load_b_frag(Wt + (nt * 16 + l15) * wStride, ks, g);
      acc[0][nt] = __builtin_amdgcn_wmma_f32_16x16x32_bf16(false, a0, false, bf,
                                                           (short)0, acc[0][nt], false, false);
      acc[1][nt] = __builtin_amdgcn_wmma_f32_16x16x32_bf16(false, a1, false, bf,
                                                           (short)0, acc[1][nt], false, false);
    }
  }
#pragma unroll
  for (int nt = 0; nt < NTILES; ++nt) {
    int col  = nt * 16 + l15;
    float bb = bias[col];
#pragma unroll
    for (int mt = 0; mt < 2; ++mt)
#pragma unroll
      for (int r = 0; r < 8; ++r) {
        int row = mt * 16 + r + 8 * g;
        float v = acc[mt][nt][r] + bb;
        v = fmaxf(v, 0.2f * v);
        Ow[row * oStride + col] = f2bf(v);
      }
  }
}

template <int KSTEPS>
__device__ inline void layer_final_half(const unsigned short* Xw, int xStride,
                                        const unsigned short* Wt, int wStride,
                                        const float* __restrict__ bias,
                                        float* __restrict__ outp, int lane) {
  const int g = lane >> 4, l15 = lane & 15;
  v8f acc[2][4];
#pragma unroll
  for (int mt = 0; mt < 2; ++mt)
#pragma unroll
    for (int nt = 0; nt < 4; ++nt) acc[mt][nt] = zero8();

#pragma unroll
  for (int ks = 0; ks < KSTEPS; ++ks) {
    v16bf a0 = load_a_frag(Xw + (0 * 16 + l15) * xStride, ks, g);
    v16bf a1 = load_a_frag(Xw + (1 * 16 + l15) * xStride, ks, g);
#pragma unroll
    for (int nt = 0; nt < 4; ++nt) {
      v16bf bf = load_b_frag(Wt + (nt * 16 + l15) * wStride, ks, g);
      acc[0][nt] = __builtin_amdgcn_wmma_f32_16x16x32_bf16(false, a0, false, bf,
                                                           (short)0, acc[0][nt], false, false);
      acc[1][nt] = __builtin_amdgcn_wmma_f32_16x16x32_bf16(false, a1, false, bf,
                                                           (short)0, acc[1][nt], false, false);
    }
  }
#pragma unroll
  for (int nt = 0; nt < 4; ++nt) {
    int col  = nt * 16 + l15;
    float bb = bias[col];
    float m  = -FLTMAX;
#pragma unroll
    for (int mt = 0; mt < 2; ++mt)
#pragma unroll
      for (int r = 0; r < 8; ++r) {
        float v = acc[mt][nt][r] + bb;
        v = fmaxf(v, 0.2f * v);
        m = fmaxf(m, v);
      }
    m = fmaxf(m, __shfl_xor(m, 16, 32));   // combine lane-group halves
    if (lane < 16) outp[col] = m;
  }
}

__global__ __launch_bounds__(128) void sa_mlp_kernel(
    const float* __restrict__ xyz, const float* __restrict__ feats,
    const float* __restrict__ b0, const float* __restrict__ b1,
    const float* __restrict__ b2,
    const unsigned short* __restrict__ wpack,
    const float* __restrict__ selxyz, const int* __restrict__ nbr,
    float* __restrict__ outfeat) {
  __shared__ __align__(16) unsigned short Wpack[WPACK_ELEMS];  // 36,864 B
  __shared__ __align__(16) unsigned short Xb[4 * 32 * 96];     // per-wave activations

  const int tid  = threadIdx.x;
  const int wave = tid >> 5, lane = tid & 31;

  // wave 0: single TDM DMA pulls the whole weight image into LDS
  if (tid < 32) {
    tdm_load_weights(wpack, (unsigned)(uintptr_t)(&Wpack[0]));
    __builtin_amdgcn_s_wait_tensorcnt(0);
  }

  const int gq = blockIdx.x * 4 + wave;  // one query per wave
  const int b  = gq >> 10;
  unsigned short* Xw = Xb + wave * (32 * 96);

  // gather: lane k builds row k = [features | local xyz | pad] (packed b32 stores)
  {
    int nid = nbr[(size_t)gq * K_ + lane];
    const float* P = xyz   + ((size_t)b * N1_ + nid) * 3;
    const float* F = feats + ((size_t)b * N1_ + nid) * F1_;
    const float* Q = selxyz + (size_t)gq * 3;
    unsigned* row32 = (unsigned*)(Xw + lane * 96);
#pragma unroll 8
    for (int c = 0; c < F1_; c += 2)
      row32[c >> 1] = (unsigned)f2bf(F[c]) | ((unsigned)f2bf(F[c + 1]) << 16);
    row32[32] = (unsigned)f2bf(P[0] - Q[0]) | ((unsigned)f2bf(P[1] - Q[1]) << 16);
    row32[33] = (unsigned)f2bf(P[2] - Q[2]);  // elem 67 = 0
#pragma unroll
    for (int k2 = 34; k2 < 48; ++k2) row32[k2] = 0u;
  }
  __syncthreads();  // weights (TDM) + activations visible

  const unsigned short* Wt0 = Wpack;
  const unsigned short* Wt1 = Wpack + WT1_OFF;
  const unsigned short* Wt2 = Wpack + WT2_OFF;

  layer_store<3, 4>(Xw, 96, Wt0, 96, b0, Xw, 96, lane);  // 32x96 @ 96x64
  __syncthreads();
  layer_store<2, 4>(Xw, 96, Wt1, 64, b1, Xw, 96, lane);  // 32x64 @ 64x64
  __syncthreads();
  float* outp = outfeat + (size_t)gq * 128;
  layer_final_half<2>(Xw, 96, Wt2,           64, b2,      outp,      lane);
  layer_final_half<2>(Xw, 96, Wt2 + 64 * 64, 64, b2 + 64, outp + 64, lane);
}

// -----------------------------------------------------------------------------
extern "C" void kernel_launch(void* const* d_in, const int* in_sizes, int n_in,
                              void* d_out, int out_size, void* d_ws, size_t ws_size,
                              hipStream_t stream) {
  (void)in_sizes; (void)n_in; (void)out_size; (void)ws_size;
  const float* xyz   = (const float*)d_in[0];
  const float* feats = (const float*)d_in[1];
  const float* W0 = (const float*)d_in[2];
  const float* b0 = (const float*)d_in[3];
  const float* W1 = (const float*)d_in[4];
  const float* b1 = (const float*)d_in[5];
  const float* W2 = (const float*)d_in[6];
  const float* b2 = (const float*)d_in[7];

  float* selxyz  = (float*)d_out;                        // (B,N2,3)
  float* outfeat = (float*)d_out + (size_t)B_ * N2_ * 3; // (B,N2,128)
  int*   nbr     = (int*)d_ws;                           // (B,N2,K) = 1 MB
  unsigned short* wpack = (unsigned short*)((char*)d_ws + (size_t)B_ * N2_ * K_ * 4);

  pack_weights_kernel<<<(WPACK_ELEMS + 255) / 256, 256, 0, stream>>>(W0, W1, W2, wpack);
  fps_kernel<<<B_, 1024, 0, stream>>>(xyz, selxyz);
  knn_kernel<<<B_ * N2_, 256, 0, stream>>>(xyz, selxyz, nbr);
  sa_mlp_kernel<<<(B_ * N2_) / 4, 128, 0, stream>>>(xyz, feats, b0, b1, b2, wpack,
                                                    selxyz, nbr, outfeat);
}
